// SetDecoderBlock_47940424958302
// MI455X (gfx1250) — matrix operand order repaired and checked by reading, not compiled, not run on hardware
//
#include <hip/hip_runtime.h>

// ---------------------------------------------------------------------------
// Transformer decoder block (self-attn + cross-attn + FFN, 3x LayerNorm) for
// MI455X / gfx1250.
//   Phase 0: one-time f32 -> bf16 conversion of inputs + all weights.
//   GEMMs:   pure bf16 operands on the WMMA pipe (v_wmma_f32_16x16x32_bf16,
//            f32 accumulate).  K-loop is software-pipelined: next panel's
//            global b128 loads are issued before the WMMA compute phase and
//            land in registers while the matrix pipe works; regs -> LDS after
//            the tail barrier.  No conversion VALU in the K-loop.
//   Outputs feeding another GEMM are written bf16; outputs feeding a
//   LayerNorm are written f32.  Mid-block LayerNorms emit dual f32+bf16.
// Shapes guarantee M % 64 == 0 and K % 64 == 0 for every GEMM; only N is
// ever guarded (P@V GEMM, N=64), selected by template.
// Workspace: ~320 MB carved from d_ws.
// ---------------------------------------------------------------------------

#define LATENT   1024
#define HIDDEN   4096
#define HEADS    16
#define HEAD_DIM 64
#define BATCH    4
#define SEQ_Q    1024
#define SEQ_K    1024

typedef __bf16 bf16;
typedef __attribute__((ext_vector_type(16))) __bf16 v16bf;
typedef __attribute__((ext_vector_type(8)))  __bf16 v8bf;
typedef __attribute__((ext_vector_type(8)))  float  v8f;
typedef __attribute__((ext_vector_type(4)))  float  v4f;
typedef __attribute__((ext_vector_type(4)))  unsigned v4u;

__device__ __forceinline__ bf16 f2bf(float f) {
  union { float f; unsigned u; } x; x.f = f;
  unsigned r = x.u + 0x7FFFu + ((x.u >> 16) & 1u);   // round-to-nearest-even
  unsigned short h = (unsigned short)(r >> 16);
  bf16 o; __builtin_memcpy(&o, &h, 2); return o;
}
__device__ __forceinline__ float bf2f(bf16 b) {
  unsigned short h; __builtin_memcpy(&h, &b, 2);
  union { unsigned u; float f; } x; x.u = ((unsigned)h) << 16; return x.f;
}
// two floats -> one dword holding two RNE-rounded bf16 (branch-free)
__device__ __forceinline__ unsigned bfpack(float a, float b) {
  union { float f; unsigned u; } xa, xb; xa.f = a; xb.f = b;
  unsigned ra = (xa.u + 0x7FFFu + ((xa.u >> 16) & 1u)) >> 16;
  unsigned rb = (xb.u + 0x7FFFu + ((xb.u >> 16) & 1u)) & 0xFFFF0000u;
  return ra | rb;
}
__device__ __forceinline__ void stC(float* p, float v) { *p = v; }
__device__ __forceinline__ void stC(bf16*  p, float v) { *p = f2bf(v); }

// ---------------------------------------------------------------------------
// One-time f32 -> bf16 conversion: 8 elements/thread, b128 in, b128 out.
// Element count is always a multiple of 2048 here (grid sized exactly).
// ---------------------------------------------------------------------------
__global__ __launch_bounds__(256)
void cvt_f32_bf16(const float* __restrict__ in, bf16* __restrict__ out)
{
  const long long i = ((long long)blockIdx.x * 256 + threadIdx.x) * 8;
  v4f a = *(const v4f*)(in + i);
  v4f b = *(const v4f*)(in + i + 4);
  v4u p;
  p[0] = bfpack(a[0], a[1]); p[1] = bfpack(a[2], a[3]);
  p[2] = bfpack(b[0], b[1]); p[3] = bfpack(b[2], b[3]);
  *(v4u*)(out + i) = p;
}

// ---------------------------------------------------------------------------
// C[M x N] = alpha * A[M x K] @ B^T (+bias[n]) (ReLU?),  A,B bf16, C = TC.
//   BKN=false : B stored (N,K) row-major (nn.Linear weight, y = x @ w^T)
//   BKN=true  : B stored (K,N) row-major (P @ V case) -> transposed LDS stage
//   GUARD     : bounds-check the N dimension only (M,K tile-aligned by data)
// Batched via blockIdx.z: z1 = z >> zshift, z2 = z & ((1<<zshift)-1).
// Block: 256 threads = 8 wave32.  Block tile 64x128, K-step 64.
// Each wave: 32x32 tile = 2x2 f32 accumulators, 8 WMMAs per K-step.
// Software-pipelined: panel k+1 global loads overlap panel k WMMA compute.
// ---------------------------------------------------------------------------
template <typename TC, bool BKN, bool RELU, bool HASBIAS, bool GUARD>
__global__ __launch_bounds__(256, 2)
void gemm_wmma_bf16(const bf16* __restrict__ Ag, const bf16* __restrict__ Bg,
                    TC* __restrict__ Cg, const float* __restrict__ bias,
                    int M, int N, int K, int lda, int ldb, int ldc,
                    long long sA1, long long sA2,
                    long long sB1, long long sB2,
                    long long sC1, long long sC2,
                    int zshift, float alpha)
{
  constexpr int BM = 64, BN = 128, BK = 64, LDK = BK + 8;  // +8 halfs: pad banks
  __shared__ __attribute__((aligned(16))) bf16 As[BM * LDK];
  __shared__ __attribute__((aligned(16))) bf16 Ws[BN * LDK];

  const int z  = blockIdx.z;
  const int z1 = z >> zshift, z2 = z & ((1 << zshift) - 1);
  const bf16* A = Ag + z1 * sA1 + z2 * sA2;
  const bf16* B = Bg + z1 * sB1 + z2 * sB2;
  TC*         C = Cg + z1 * sC1 + z2 * sC2;

  const int m0 = blockIdx.y * BM;
  const int n0 = blockIdx.x * BN;

  const int tid  = threadIdx.x;
  const int lane = tid & 31;
  const int wave = tid >> 5;
  const int wm   = (wave >> 2) * 32;  // 0 | 32
  const int wn   = (wave & 3)  * 32;  // 0 | 32 | 64 | 96

  v8f acc[2][2];
#pragma unroll
  for (int i = 0; i < 2; ++i)
#pragma unroll
    for (int j = 0; j < 2; ++j)
#pragma unroll
      for (int r = 0; r < 8; ++r) acc[i][j][r] = 0.0f;

  // ---- register-buffered staging: 2 A chunks + 4 B chunks per thread ----
  v8bf ra[2], rb[4];

  auto load_tiles = [&](int kk) {
#pragma unroll
    for (int it = 0; it < 2; ++it) {
      int chunk = it * 256 + tid;
      int ml = chunk >> 3;
      int kc = (chunk & 7) * 8;
      ra[it] = *(const v8bf*)&A[(long long)(m0 + ml) * lda + kk + kc];
    }
    if (!BKN) {
#pragma unroll
      for (int it = 0; it < 4; ++it) {
        int chunk = it * 256 + tid;
        int nl = chunk >> 3;
        int kc = (chunk & 7) * 8;
        int gn = n0 + nl;
        v8bf v; __builtin_memset(&v, 0, sizeof(v));
        if (!GUARD || gn < N) v = *(const v8bf*)&B[(long long)gn * ldb + kk + kc];
        rb[it] = v;
      }
    } else {  // B is (K,N): coalesced b128 over n
#pragma unroll
      for (int it = 0; it < 4; ++it) {
        int chunk = it * 256 + tid;
        int kl = chunk >> 4;
        int nc = (chunk & 15) * 8;
        int gn = n0 + nc;                       // N % 8 == 0 always here
        v8bf v; __builtin_memset(&v, 0, sizeof(v));
        if (!GUARD || gn < N) v = *(const v8bf*)&B[(long long)(kk + kl) * ldb + gn];
        rb[it] = v;
      }
    }
  };

  auto store_tiles = [&]() {
#pragma unroll
    for (int it = 0; it < 2; ++it) {
      int chunk = it * 256 + tid;
      int ml = chunk >> 3;
      int kc = (chunk & 7) * 8;
      *(v8bf*)&As[ml * LDK + kc] = ra[it];
    }
    if (!BKN) {
#pragma unroll
      for (int it = 0; it < 4; ++it) {
        int chunk = it * 256 + tid;
        int nl = chunk >> 3;
        int kc = (chunk & 7) * 8;
        *(v8bf*)&Ws[nl * LDK + kc] = rb[it];
      }
    } else {  // transpose into Ws[n][k]
#pragma unroll
      for (int it = 0; it < 4; ++it) {
        int chunk = it * 256 + tid;
        int kl = chunk >> 4;
        int nc = (chunk & 15) * 8;
#pragma unroll
        for (int j = 0; j < 8; ++j) Ws[(nc + j) * LDK + kl] = rb[it][j];
      }
    }
  };

  load_tiles(0);
  store_tiles();

  for (int k0 = 0; k0 < K; k0 += BK) {
    __syncthreads();                       // LDS tile for panel k0 is ready
    const bool more = (k0 + BK) < K;
    if (more) load_tiles(k0 + BK);         // global loads fly under compute

    // ---- compute: two 16x16x32 sub-steps per K-panel ----
#pragma unroll
    for (int ks = 0; ks < BK; ks += 32) {
      v16bf af[2], bg[2];
      // A frag (16x32 bf16): lanes 0-15 rows m, K 0-7 & 16-23; lanes 16-31 K 8-15 & 24-31
      const int am  = lane & 15;
      const int akb = (lane >> 4) * 8;
#pragma unroll
      for (int mi = 0; mi < 2; ++mi) {
        const bf16* base = &As[(wm + mi * 16 + am) * LDK + ks];
        v8bf lo = *(const v8bf*)(base + akb);
        v8bf hi = *(const v8bf*)(base + 16 + akb);
        af[mi] = __builtin_shufflevector(lo, hi, 0,1,2,3,4,5,6,7,8,9,10,11,12,13,14,15);
      }
      // B frag (32x16 bf16): lane n, lanes 0-15 K 0-7 & 16-23; lanes 16-31 K 8-15 & 24-31
      const int bn  = lane & 15;
      const int bkb = (lane >> 4) * 8;
#pragma unroll
      for (int ni = 0; ni < 2; ++ni) {
        const bf16* base = &Ws[(wn + ni * 16 + bn) * LDK + ks + bkb];
        v8bf lo = *(const v8bf*)(base);
        v8bf hi = *(const v8bf*)(base + 16);
        bg[ni] = __builtin_shufflevector(lo, hi, 0,1,2,3,4,5,6,7,8,9,10,11,12,13,14,15);
      }
#pragma unroll
      for (int mi = 0; mi < 2; ++mi)
#pragma unroll
        for (int ni = 0; ni < 2; ++ni)
          acc[mi][ni] = __builtin_amdgcn_wmma_f32_16x16x32_bf16(
              false, af[mi], false, bg[ni], (short)0, acc[mi][ni], false, false);
    }

    __syncthreads();                       // all waves done reading LDS
    if (more) store_tiles();               // regs -> LDS for panel k0+BK
  }

  // ---- epilogue: C/D layout: VGPR r -> M=(lane>>4)*8+r, N=lane&15 ----
  const int cm = (lane >> 4) * 8;
  const int cn = lane & 15;
#pragma unroll
  for (int mi = 0; mi < 2; ++mi) {
#pragma unroll
    for (int ni = 0; ni < 2; ++ni) {
      int gn = n0 + wn + ni * 16 + cn;
      if (GUARD && gn >= N) continue;
      float bv = 0.0f;
      if (HASBIAS) bv = bias[gn];
#pragma unroll
      for (int r = 0; r < 8; ++r) {
        int gm = m0 + wm + mi * 16 + cm + r;     // M % 64 == 0: never OOB
        float v = acc[mi][ni][r] * alpha + bv;
        if (RELU) v = fmaxf(v, 0.0f);
        stC(&C[(long long)gm * ldc + gn], v);
      }
    }
  }
}

// ---------------------------------------------------------------------------
// In-place row softmax on bf16 scores; one block per row, ncols must be 1024.
// ---------------------------------------------------------------------------
__global__ __launch_bounds__(256)
void softmax_rows_bf16(bf16* __restrict__ S, int ncols)
{
  __shared__ float red[256];
  const long long row = blockIdx.x;
  bf16* p = S + row * (long long)ncols;
  const int tid = threadIdx.x;
  float v[4];
  float mx = -3.402823466e38f;
#pragma unroll
  for (int i = 0; i < 4; ++i) { v[i] = bf2f(p[tid + i * 256]); mx = fmaxf(mx, v[i]); }
  red[tid] = mx; __syncthreads();
  for (int s = 128; s > 0; s >>= 1) { if (tid < s) red[tid] = fmaxf(red[tid], red[tid + s]); __syncthreads(); }
  const float rmax = red[0]; __syncthreads();
  float sum = 0.0f;
#pragma unroll
  for (int i = 0; i < 4; ++i) { v[i] = __expf(v[i] - rmax); sum += v[i]; }
  red[tid] = sum; __syncthreads();
  for (int s = 128; s > 0; s >>= 1) { if (tid < s) red[tid] += red[tid + s]; __syncthreads(); }
  const float inv = 1.0f / red[0];
#pragma unroll
  for (int i = 0; i < 4; ++i) p[tid + i * 256] = f2bf(v[i] * inv);
}

// ---------------------------------------------------------------------------
// out = LayerNorm(X + Y) * g + b, row length LATENT; one block per row.
// DUAL=true additionally writes a bf16 copy (GEMM-A operand for next stage).
// ---------------------------------------------------------------------------
template <bool DUAL>
__global__ __launch_bounds__(256)
void add_layernorm(const float* __restrict__ X, const float* __restrict__ Y,
                   const float* __restrict__ g, const float* __restrict__ b,
                   float* __restrict__ out, bf16* __restrict__ out_bf)
{
  __shared__ float red[256];
  const long long row = blockIdx.x;
  const float* px = X + row * LATENT;
  const float* py = Y + row * LATENT;
  float* po = out + row * LATENT;
  bf16* pb = DUAL ? (out_bf + row * LATENT) : nullptr;
  const int tid = threadIdx.x;
  float v[4]; float s = 0.0f, s2 = 0.0f;
#pragma unroll
  for (int i = 0; i < 4; ++i) {
    int c = tid + i * 256;
    v[i] = px[c] + py[c];
    s += v[i]; s2 += v[i] * v[i];
  }
  red[tid] = s; __syncthreads();
  for (int t = 128; t > 0; t >>= 1) { if (tid < t) red[tid] += red[tid + t]; __syncthreads(); }
  const float mean = red[0] * (1.0f / LATENT); __syncthreads();
  red[tid] = s2; __syncthreads();
  for (int t = 128; t > 0; t >>= 1) { if (tid < t) red[tid] += red[tid + t]; __syncthreads(); }
  const float var = red[0] * (1.0f / LATENT) - mean * mean;
  const float inv = rsqrtf(var + 1e-5f);
#pragma unroll
  for (int i = 0; i < 4; ++i) {
    int c = tid + i * 256;
    float r = (v[i] - mean) * inv * g[c] + b[c];
    po[c] = r;
    if (DUAL) pb[c] = f2bf(r);
  }
}

// ---------------------------------------------------------------------------
extern "C" void kernel_launch(void* const* d_in, const int* in_sizes, int n_in,
                              void* d_out, int out_size, void* d_ws, size_t ws_size,
                              hipStream_t stream)
{
  const float* Q     = (const float*)d_in[0];
  const float* Kenc  = (const float*)d_in[1];
  const float* a1_wq = (const float*)d_in[2];  const float* a1_bq = (const float*)d_in[3];
  const float* a1_wk = (const float*)d_in[4];  const float* a1_bk = (const float*)d_in[5];
  const float* a1_wv = (const float*)d_in[6];  const float* a1_bv = (const float*)d_in[7];
  const float* a1_wo = (const float*)d_in[8];  const float* a1_bo = (const float*)d_in[9];
  const float* a2_wq = (const float*)d_in[10]; const float* a2_bq = (const float*)d_in[11];
  const float* a2_wk = (const float*)d_in[12]; const float* a2_bk = (const float*)d_in[13];
  const float* a2_wv = (const float*)d_in[14]; const float* a2_bv = (const float*)d_in[15];
  const float* a2_wo = (const float*)d_in[16]; const float* a2_bo = (const float*)d_in[17];
  const float* fc_w1 = (const float*)d_in[18]; const float* fc_b1 = (const float*)d_in[19];
  const float* fc_w2 = (const float*)d_in[20]; const float* fc_b2 = (const float*)d_in[21];
  const float* ln0_g = (const float*)d_in[22]; const float* ln0_b = (const float*)d_in[23];
  const float* ln1_g = (const float*)d_in[24]; const float* ln1_b = (const float*)d_in[25];
  const float* ln2_g = (const float*)d_in[26]; const float* ln2_b = (const float*)d_in[27];

  const size_t MB = 1024ull * 1024ull;
  char* ws = (char*)d_ws;
  bf16* Qbf    = (bf16*)(ws +   0 * MB);  //  8 MB (B*SQ, LATENT)
  bf16* Kencbf = (bf16*)(ws +   8 * MB);  //  8 MB
  bf16* w1q = (bf16*)(ws + 16 * MB);  bf16* w1k = (bf16*)(ws + 18 * MB);  // 2 MB each
  bf16* w1v = (bf16*)(ws + 20 * MB);  bf16* w1o = (bf16*)(ws + 22 * MB);
  bf16* w2q = (bf16*)(ws + 24 * MB);  bf16* w2k = (bf16*)(ws + 26 * MB);
  bf16* w2v = (bf16*)(ws + 28 * MB);  bf16* w2o = (bf16*)(ws + 30 * MB);
  bf16* wf1 = (bf16*)(ws + 32 * MB);  //  8 MB (HIDDEN, LATENT)
  bf16* wf2 = (bf16*)(ws + 40 * MB);  //  8 MB (LATENT, HIDDEN)
  bf16* Qp  = (bf16*)(ws + 48 * MB);  //  8 MB
  bf16* Kp  = (bf16*)(ws + 56 * MB);  //  8 MB
  bf16* Vp  = (bf16*)(ws + 64 * MB);  //  8 MB
  bf16* CTX = (bf16*)(ws + 72 * MB);  //  8 MB
  float* O   = (float*)(ws +  80 * MB);  // 16 MB
  float* X1f = (float*)(ws +  96 * MB);  // 16 MB
  bf16*  X1b = (bf16*) (ws + 112 * MB);  //  8 MB
  float* X2f = (float*)(ws + 120 * MB);  // 16 MB
  bf16*  X2b = (bf16*) (ws + 136 * MB);  //  8 MB
  bf16*  Hbf = (bf16*) (ws + 144 * MB);  // 32 MB (B*SQ, HIDDEN)
  float* Fb  = (float*)(ws + 176 * MB);  // 16 MB
  bf16*  S   = (bf16*) (ws + 192 * MB);  // 128 MB -> 320 MB total

  const int  Mrow = BATCH * SEQ_Q;           // 4096
  const float scl = 0.03125f;                // 1/sqrt(LATENT)
  dim3 blk(256);
  #define GRID(M_, N_, Z_) dim3((unsigned)(((N_) + 127) / 128), (unsigned)(((M_) + 63) / 64), (unsigned)(Z_))
  #define CVT(src, dst, nelem) cvt_f32_bf16<<<dim3((unsigned)((nelem) / 2048)), blk, 0, stream>>>(src, dst)

  // ===== phase 0: one-time f32 -> bf16 conversions =====
  CVT(Q,     Qbf,    (long long)Mrow * LATENT);
  CVT(Kenc,  Kencbf, (long long)Mrow * LATENT);
  CVT(a1_wq, w1q, LATENT * LATENT);  CVT(a1_wk, w1k, LATENT * LATENT);
  CVT(a1_wv, w1v, LATENT * LATENT);  CVT(a1_wo, w1o, LATENT * LATENT);
  CVT(a2_wq, w2q, LATENT * LATENT);  CVT(a2_wk, w2k, LATENT * LATENT);
  CVT(a2_wv, w2v, LATENT * LATENT);  CVT(a2_wo, w2o, LATENT * LATENT);
  CVT(fc_w1, wf1, HIDDEN * LATENT);  CVT(fc_w2, wf2, LATENT * HIDDEN);

  const long long sQb = (long long)SEQ_Q * LATENT;   // per-batch stride in proj
  const long long sSb = (long long)HEADS * SEQ_Q * SEQ_K;
  const long long sSh = (long long)SEQ_Q * SEQ_K;

  // ===================== self-attention =====================
  gemm_wmma_bf16<bf16, false, false, true, false><<<GRID(Mrow, LATENT, 1), blk, 0, stream>>>(
      Qbf, w1q, Qp, a1_bq, Mrow, LATENT, LATENT, LATENT, LATENT, LATENT, 0,0,0,0,0,0, 0, 1.0f);
  gemm_wmma_bf16<bf16, false, false, true, false><<<GRID(Mrow, LATENT, 1), blk, 0, stream>>>(
      Qbf, w1k, Kp, a1_bk, Mrow, LATENT, LATENT, LATENT, LATENT, LATENT, 0,0,0,0,0,0, 0, 1.0f);
  gemm_wmma_bf16<bf16, false, false, true, false><<<GRID(Mrow, LATENT, 1), blk, 0, stream>>>(
      Qbf, w1v, Vp, a1_bv, Mrow, LATENT, LATENT, LATENT, LATENT, LATENT, 0,0,0,0,0,0, 0, 1.0f);

  // S[b,h] = scl * Q_h @ K_h^T   (batched over b*HEADS, zshift=4)
  gemm_wmma_bf16<bf16, false, false, false, false><<<GRID(SEQ_Q, SEQ_K, BATCH * HEADS), blk, 0, stream>>>(
      Qp, Kp, S, nullptr, SEQ_Q, SEQ_K, HEAD_DIM, LATENT, LATENT, SEQ_K,
      sQb, HEAD_DIM, sQb, HEAD_DIM, sSb, sSh, 4, scl);
  softmax_rows_bf16<<<dim3(BATCH * HEADS * SEQ_Q), blk, 0, stream>>>(S, SEQ_K);
  // CTX[b,q,h*64+d] = P_h @ V_h   (B in K x N layout, N=64 -> GUARD)
  gemm_wmma_bf16<bf16, true, false, false, true><<<GRID(SEQ_Q, HEAD_DIM, BATCH * HEADS), blk, 0, stream>>>(
      S, Vp, CTX, nullptr, SEQ_Q, HEAD_DIM, SEQ_K, SEQ_K, LATENT, LATENT,
      sSb, sSh, sQb, HEAD_DIM, sQb, HEAD_DIM, 4, 1.0f);

  gemm_wmma_bf16<float, false, false, true, false><<<GRID(Mrow, LATENT, 1), blk, 0, stream>>>(
      CTX, w1o, O, a1_bo, Mrow, LATENT, LATENT, LATENT, LATENT, LATENT, 0,0,0,0,0,0, 0, 1.0f);
  add_layernorm<true><<<dim3(Mrow), blk, 0, stream>>>(Q, O, ln0_g, ln0_b, X1f, X1b);

  // ===================== cross-attention =====================
  gemm_wmma_bf16<bf16, false, false, true, false><<<GRID(Mrow, LATENT, 1), blk, 0, stream>>>(
      X1b, w2q, Qp, a2_bq, Mrow, LATENT, LATENT, LATENT, LATENT, LATENT, 0,0,0,0,0,0, 0, 1.0f);
  gemm_wmma_bf16<bf16, false, false, true, false><<<GRID(Mrow, LATENT, 1), blk, 0, stream>>>(
      Kencbf, w2k, Kp, a2_bk, Mrow, LATENT, LATENT, LATENT, LATENT, LATENT, 0,0,0,0,0,0, 0, 1.0f);
  gemm_wmma_bf16<bf16, false, false, true, false><<<GRID(Mrow, LATENT, 1), blk, 0, stream>>>(
      Kencbf, w2v, Vp, a2_bv, Mrow, LATENT, LATENT, LATENT, LATENT, LATENT, 0,0,0,0,0,0, 0, 1.0f);

  gemm_wmma_bf16<bf16, false, false, false, false><<<GRID(SEQ_Q, SEQ_K, BATCH * HEADS), blk, 0, stream>>>(
      Qp, Kp, S, nullptr, SEQ_Q, SEQ_K, HEAD_DIM, LATENT, LATENT, SEQ_K,
      sQb, HEAD_DIM, sQb, HEAD_DIM, sSb, sSh, 4, scl);
  softmax_rows_bf16<<<dim3(BATCH * HEADS * SEQ_Q), blk, 0, stream>>>(S, SEQ_K);
  gemm_wmma_bf16<bf16, true, false, false, true><<<GRID(SEQ_Q, HEAD_DIM, BATCH * HEADS), blk, 0, stream>>>(
      S, Vp, CTX, nullptr, SEQ_Q, HEAD_DIM, SEQ_K, SEQ_K, LATENT, LATENT,
      sSb, sSh, sQb, HEAD_DIM, sQb, HEAD_DIM, 4, 1.0f);

  gemm_wmma_bf16<float, false, false, true, false><<<GRID(Mrow, LATENT, 1), blk, 0, stream>>>(
      CTX, w2o, O, a2_bo, Mrow, LATENT, LATENT, LATENT, LATENT, LATENT, 0,0,0,0,0,0, 0, 1.0f);
  add_layernorm<true><<<dim3(Mrow), blk, 0, stream>>>(X1f, O, ln1_g, ln1_b, X2f, X2b);

  // ===================== FFN =====================
  gemm_wmma_bf16<bf16, false, true, true, false><<<GRID(Mrow, HIDDEN, 1), blk, 0, stream>>>(
      X2b, wf1, Hbf, fc_b1, Mrow, HIDDEN, LATENT, LATENT, LATENT, HIDDEN, 0,0,0,0,0,0, 0, 1.0f);
  gemm_wmma_bf16<float, false, false, true, false><<<GRID(Mrow, LATENT, 1), blk, 0, stream>>>(
      Hbf, wf2, Fb, fc_b2, Mrow, LATENT, HIDDEN, HIDDEN, HIDDEN, LATENT, 0,0,0,0,0,0, 0, 1.0f);
  add_layernorm<false><<<dim3(Mrow), blk, 0, stream>>>(X2f, Fb, ln2_g, ln2_b, (float*)d_out, nullptr);

  #undef GRID
  #undef CVT
}
